// _CGCNN_49383533969986
// MI455X (gfx1250) — compile-verified, matrix operand order checked by reading, not run on hardware
//
#include <hip/hip_runtime.h>
#include <hip/hip_bf16.h>

typedef __attribute__((ext_vector_type(16))) _Float16 v16h;
typedef __attribute__((ext_vector_type(8)))  _Float16 v8h;
typedef __attribute__((ext_vector_type(4)))  _Float16 v4h;
typedef __attribute__((ext_vector_type(8)))  float    v8f;

#define N_NODES  100000
#define N_EDGES  1000000
#define N_GRAPHS 512
#define HID      64
#define EDIM     32
#define LAYERS   4
#define ZW       160      // 2*HID + EDIM
#define BN_EPS   1e-5f
#define KCHUNKS  5        // 160 / 32
#define NTILES   4        // 64 / 16
#define TPW      8        // 16-edge tiles per wave (1024 edges per block)
#define WTILE_HALVES (2 * KCHUNKS * NTILES * 512)   // 20480 halves / layer (f + s)

__device__ __forceinline__ float fast_sigmoid(float v) {
    return __builtin_amdgcn_rcpf(1.0f + __expf(-v));
}
__device__ __forceinline__ float fast_softplus(float v) {
    // v_exp_f32 + v_log_f32; arg of log >= 1 so no cancellation issue
    float sp = __logf(1.0f + __expf(v));
    return (v > 20.0f) ? v : sp;
}

// ---------------------------------------------------------------------------
// Utility: zero a float buffer
// ---------------------------------------------------------------------------
__global__ void zerof_kernel(float* __restrict__ p, long n) {
    long i = (long)blockIdx.x * blockDim.x + threadIdx.x;
    if (i < n) p[i] = 0.0f;
}

// ---------------------------------------------------------------------------
// Embedding gather (float4-wide): x[n][c] = emb[z[n]][c]
// ---------------------------------------------------------------------------
__global__ void embed_kernel(const int* __restrict__ z, const float* __restrict__ emb,
                             float* __restrict__ x) {
    long idx = (long)blockIdx.x * blockDim.x + threadIdx.x;   // quad index
    if (idx >= (long)N_NODES * (HID / 4)) return;
    long n = idx >> 4;
    int  q = idx & 15;
    ((float4*)x)[idx] = ((const float4*)emb)[(long)z[n] * (HID / 4) + q];
}

// ---------------------------------------------------------------------------
// Repack Wf/Ws (f32 [L][160][64]) into per-lane WMMA B-fragments (f16).
// Tile index: ((l*2 + w)*KCHUNKS + c)*NTILES + nt ; each tile = 32 lanes x 16 halves.
// B layout (16-bit, 32x16): lane<16 holds col N=lane, halves i<8 -> K=c*32+i,
// i>=8 -> K=c*32+16+(i-8); lanes>=16 same with +8 K offset, col = lane-16.
// ---------------------------------------------------------------------------
__global__ void repack_weights_kernel(const float* __restrict__ Wf,
                                      const float* __restrict__ Ws,
                                      _Float16* __restrict__ wh) {
    int gid = blockIdx.x * blockDim.x + threadIdx.x;
    if (gid >= LAYERS * 2 * KCHUNKS * NTILES * 32) return;
    int lane = gid & 31;
    int t    = gid >> 5;
    int nt   = t & 3;  t >>= 2;
    int c    = t % KCHUNKS; t /= KCHUNKS;
    int w    = t & 1;
    int l    = t >> 1;
    const float* W = (w ? Ws : Wf) + (long)l * ZW * HID;
    int n    = nt * 16 + (lane & 15);
    int koff = (lane < 16) ? 0 : 8;
#pragma unroll
    for (int i = 0; i < 16; ++i) {
        int kk = koff + (i < 8 ? i : i + 8);
        int k  = c * 32 + kk;
        wh[(long)gid * 16 + i] = (_Float16)W[(long)k * HID + n];
    }
}

// ---------------------------------------------------------------------------
// Edge message kernel. Per block: stage this layer's 40KB of B-fragments into
// LDS once; each of 8 waves then processes TPW 16-edge tiles:
//   gather z_e = [x_dst | x_src | e] (float4 loads -> f16 LDS tile),
//   5x4x2 V_WMMA_F32_16X16X32_F16 with LDS A and B fragments,
//   sigmoid*softplus epilogue, global_atomic_add_f32 scatter by dst.
// Tile loop and K-chunk loop are kept rolled (#pragma unroll 1) so B fragments
// stay in LDS instead of being register-cached -> moderate VGPR use, high
// occupancy for the gather/atomic-bound critical path.
// ---------------------------------------------------------------------------
__global__ __launch_bounds__(256)
void edge_msg_kernel(const float* __restrict__ x,
                     const int* __restrict__ eidx,       // [2][E]: row0=src, row1=dst
                     const float* __restrict__ eattr,    // [E][EDIM]
                     const _Float16* __restrict__ wh,    // repacked weights
                     const float* __restrict__ bf,       // [L][HID]
                     const float* __restrict__ bs,       // [L][HID]
                     float* __restrict__ agg,            // [N][HID]
                     int layer) {
    __shared__ _Float16 wlds[WTILE_HALVES];   // 40 KB: f tiles [0,20) then s tiles [20,40)
    __shared__ _Float16 zt[8][16 * ZW];       // 8 x 5 KB
    __shared__ int dsts[8][16];
    __shared__ int srcs[8][16];

    const int lane = threadIdx.x & 31;
    const int wv   = threadIdx.x >> 5;

    // ---- Stage this layer's weight fragments into LDS (once per block) ----
    {
        const uint4* src = (const uint4*)(wh + (long)layer * WTILE_HALVES);
        uint4* dst = (uint4*)wlds;
        for (int i = threadIdx.x; i < WTILE_HALVES / 8; i += 256)  // 2560 x 16B
            dst[i] = src[i];
    }
    __syncthreads();

    const int col  = lane & 15;
    const int koff = (lane < 16) ? 0 : 8;
    const int* srcI = eidx;
    const int* dstI = eidx + N_EDGES;

    float vbf[NTILES], vbs[NTILES];
#pragma unroll
    for (int nt = 0; nt < NTILES; ++nt) {
        vbf[nt] = bf[layer * HID + nt * 16 + col];
        vbs[nt] = bs[layer * HID + nt * 16 + col];
    }

#pragma unroll 1
    for (int t = 0; t < TPW; ++t) {
        const long tile = (long)blockIdx.x * (8 * TPW) + wv * TPW + t;
        const long e0   = tile * 16;

        if (lane < 16) {
            long e = e0 + lane;
            dsts[wv][lane] = (e < N_EDGES) ? dstI[e] : -1;
            srcs[wv][lane] = (e < N_EDGES) ? srcI[e] : -1;
        }
        // same-wave LDS ops are in-order: no barrier needed (private region)

        // ---- Gather the 16x160 z_e tile (float4 granularity) ----
#pragma unroll
        for (int i = 0; i < 20; ++i) {
            int q = i * 32 + lane;        // quad index in [0, 640)
            int r = q / 40;
            int p = q % 40;
            long e = e0 + r;
            float4 v = make_float4(0.f, 0.f, 0.f, 0.f);
            if (e < N_EDGES) {
                if (p < 16)
                    v = *(const float4*)&x[(long)dsts[wv][r] * HID + p * 4];
                else if (p < 32)
                    v = *(const float4*)&x[(long)srcs[wv][r] * HID + (p - 16) * 4];
                else
                    v = *(const float4*)&eattr[e * EDIM + (p - 32) * 4];
            }
            v4h h;
            h[0] = (_Float16)v.x; h[1] = (_Float16)v.y;
            h[2] = (_Float16)v.z; h[3] = (_Float16)v.w;
            *(v4h*)&zt[wv][r * ZW + p * 4] = h;
        }

        // ---- WMMA: acc[nt] = bias; acc += A(c) x B(c,nt) over 5 K-chunks ----
        v8f accF[NTILES], accS[NTILES];
#pragma unroll
        for (int nt = 0; nt < NTILES; ++nt) {
            float vf = vbf[nt], vs = vbs[nt];
            accF[nt] = (v8f){vf, vf, vf, vf, vf, vf, vf, vf};
            accS[nt] = (v8f){vs, vs, vs, vs, vs, vs, vs, vs};
        }

#pragma unroll 1
        for (int c = 0; c < KCHUNKS; ++c) {
            const _Float16* rp = &zt[wv][(lane & 15) * ZW + c * 32 + koff];
            v8h lo = *(const v8h*)rp;
            v8h hi = *(const v8h*)(rp + 16);
            v16h a;
#pragma unroll
            for (int i = 0; i < 8; ++i) { a[i] = lo[i]; a[8 + i] = hi[i]; }

#pragma unroll
            for (int nt = 0; nt < NTILES; ++nt) {
                const _Float16* bp = &wlds[(c * NTILES + nt) * 512 + lane * 16];
                v16h bF = *(const v16h*)bp;
                v16h bS = *(const v16h*)(bp + KCHUNKS * NTILES * 512);
                accF[nt] = __builtin_amdgcn_wmma_f32_16x16x32_f16(
                    false, a, false, bF, (short)0, accF[nt], false, false);
                accS[nt] = __builtin_amdgcn_wmma_f32_16x16x32_f16(
                    false, a, false, bS, (short)0, accS[nt], false, false);
            }
        }

        // ---- Epilogue: msg = sigmoid(f)*softplus(s); scatter-add by dst ----
        // r-outer / nt-inner: one LDS dst load + one row base per 4 atomics.
        const int rbase = (lane < 16) ? 0 : 8;
#pragma unroll
        for (int r = 0; r < 8; ++r) {
            int d = dsts[wv][rbase + r];
            if (d >= 0) {
                float* ap = &agg[(long)d * HID + col];
#pragma unroll
                for (int nt = 0; nt < NTILES; ++nt) {
                    float msg = fast_sigmoid(accF[nt][r]) * fast_softplus(accS[nt][r]);
                    atomicAdd(ap + nt * 16, msg);
                }
            }
        }
    }
}

// ---------------------------------------------------------------------------
// BatchNorm statistics: per-column sum and sum-of-squares into stats[0:64],
// stats[64:128] (must be pre-zeroed). 512 rows per block.
// ---------------------------------------------------------------------------
__global__ __launch_bounds__(256)
void bn_stats_kernel(const float* __restrict__ v, float* __restrict__ stats) {
    __shared__ float pS[256], pQ[256];
    int c   = threadIdx.x & 63;
    int sub = threadIdx.x >> 6;
    long r0   = (long)blockIdx.x * 512 + sub;
    long rend = (long)(blockIdx.x + 1) * 512;
    if (rend > N_NODES) rend = N_NODES;
    float s = 0.0f, q = 0.0f;
    for (long r = r0; r < rend; r += 4) {
        float t = v[r * HID + c];
        s += t; q += t * t;
    }
    pS[threadIdx.x] = s;
    pQ[threadIdx.x] = q;
    __syncthreads();
    if (sub == 0) {
        s = pS[c] + pS[64 + c] + pS[128 + c] + pS[192 + c];
        q = pQ[c] + pQ[64 + c] + pQ[128 + c] + pQ[192 + c];
        atomicAdd(&stats[c], s);
        atomicAdd(&stats[64 + c], q);
    }
}

// ---------------------------------------------------------------------------
// agg = cbn(agg) + x   (in place on agg)
// ---------------------------------------------------------------------------
__global__ void apply_cbn_res_kernel(float* __restrict__ agg, const float* __restrict__ x,
                                     const float* __restrict__ stats,
                                     const float* __restrict__ g, const float* __restrict__ b) {
    long idx = (long)blockIdx.x * blockDim.x + threadIdx.x;
    if (idx >= (long)N_NODES * HID) return;
    int   c     = idx & 63;
    float inv_n = 1.0f / (float)N_NODES;
    float mean  = stats[c] * inv_n;
    float var   = stats[64 + c] * inv_n - mean * mean;
    float t     = (agg[idx] - mean) * rsqrtf(var + BN_EPS) * g[c] + b[c] + x[idx];
    agg[idx] = t;
}

// ---------------------------------------------------------------------------
// x = softplus(obn(t))
// ---------------------------------------------------------------------------
__global__ void apply_obn_sp_kernel(const float* __restrict__ t, float* __restrict__ x,
                                    const float* __restrict__ stats,
                                    const float* __restrict__ g, const float* __restrict__ b) {
    long idx = (long)blockIdx.x * blockDim.x + threadIdx.x;
    if (idx >= (long)N_NODES * HID) return;
    int   c     = idx & 63;
    float inv_n = 1.0f / (float)N_NODES;
    float mean  = stats[c] * inv_n;
    float var   = stats[64 + c] * inv_n - mean * mean;
    float u     = (t[idx] - mean) * rsqrtf(var + BN_EPS) * g[c] + b[c];
    x[idx] = fast_softplus(u);
}

// ---------------------------------------------------------------------------
// Global mean-pool segment sums
// ---------------------------------------------------------------------------
__global__ void pool_kernel(const float* __restrict__ x, const int* __restrict__ batch_ids,
                            float* __restrict__ pooled, float* __restrict__ cnts) {
    long idx = (long)blockIdx.x * blockDim.x + threadIdx.x;
    if (idx >= (long)N_NODES * HID) return;
    long n = idx >> 6;
    int  c = idx & 63;
    int  b = batch_ids[n];
    atomicAdd(&pooled[(long)b * HID + c], x[idx]);
    if (c == 0) atomicAdd(&cnts[b], 1.0f);
}

// ---------------------------------------------------------------------------
// Final MLP head: out[g] = silu(mean_pool @ W1 + b1) @ W2 + b2
// ---------------------------------------------------------------------------
__global__ void mlp_kernel(const float* __restrict__ pooled, const float* __restrict__ cnts,
                           const float* __restrict__ W1, const float* __restrict__ b1,
                           const float* __restrict__ W2, const float* __restrict__ b2,
                           float* __restrict__ out) {
    int g = blockIdx.x * blockDim.x + threadIdx.x;
    if (g >= N_GRAPHS) return;
    float row[HID];
    float cn = cnts[g];
    if (cn < 1.0f) cn = 1.0f;
#pragma unroll
    for (int k = 0; k < HID; ++k) row[k] = pooled[(long)g * HID + k] / cn;
    float o = b2[0];
    for (int j = 0; j < HID; ++j) {
        float a = b1[j];
#pragma unroll
        for (int k = 0; k < HID; ++k) a += row[k] * W1[k * HID + j];
        o += a * fast_sigmoid(a) * W2[j];   // silu(a) * W2
    }
    out[g] = o;
}

// ---------------------------------------------------------------------------
// Host launcher
// ---------------------------------------------------------------------------
extern "C" void kernel_launch(void* const* d_in, const int* in_sizes, int n_in,
                              void* d_out, int out_size, void* d_ws, size_t ws_size,
                              hipStream_t stream) {
    const int*   z         = (const int*)  d_in[0];
    const int*   eidx      = (const int*)  d_in[1];
    const float* eattr     = (const float*)d_in[2];
    const int*   batch_ids = (const int*)  d_in[3];
    const float* emb       = (const float*)d_in[4];
    const float* Wf        = (const float*)d_in[5];
    const float* bfv       = (const float*)d_in[6];
    const float* Ws        = (const float*)d_in[7];
    const float* bsv       = (const float*)d_in[8];
    const float* g_cbn     = (const float*)d_in[9];
    const float* b_cbn     = (const float*)d_in[10];
    const float* g_obn     = (const float*)d_in[11];
    const float* b_obn     = (const float*)d_in[12];
    const float* W1        = (const float*)d_in[13];
    const float* b1        = (const float*)d_in[14];
    const float* W2        = (const float*)d_in[15];
    const float* b2        = (const float*)d_in[16];
    float* out = (float*)d_out;

    // Workspace layout (all 256B-aligned)
    char* ws = (char*)d_ws;
    const long XBYTES = (long)N_NODES * HID * sizeof(float);   // 25,600,000
    float*    x      = (float*)ws;
    float*    agg    = (float*)(ws + XBYTES);
    _Float16* wh     = (_Float16*)(ws + 2 * XBYTES);           // 163,840 B
    float*    stats1 = (float*)(ws + 2 * XBYTES + 163840);
    float*    stats2 = (float*)(ws + 2 * XBYTES + 163840 + 512);
    float*    pooled = (float*)(ws + 2 * XBYTES + 163840 + 1024);
    float*    cnts   = (float*)(ws + 2 * XBYTES + 163840 + 1024 + (long)N_GRAPHS * HID * sizeof(float));

    const long NELT   = (long)N_NODES * HID;
    const int  EB     = (NELT + 255) / 256;                        // elementwise blocks
    const int  EDGE_B = (N_EDGES + 8 * TPW * 16 - 1) / (8 * TPW * 16);  // 1024 edges/block
    const int  STAT_B = (N_NODES + 511) / 512;

    repack_weights_kernel<<<(LAYERS * 2 * KCHUNKS * NTILES * 32 + 255) / 256, 256, 0, stream>>>(Wf, Ws, wh);
    embed_kernel<<<(int)(((long)N_NODES * (HID / 4) + 255) / 256), 256, 0, stream>>>(z, emb, x);

    for (int l = 0; l < LAYERS; ++l) {
        zerof_kernel<<<EB, 256, 0, stream>>>(agg, NELT);
        zerof_kernel<<<1, 256, 0, stream>>>(stats1, 128);
        zerof_kernel<<<1, 256, 0, stream>>>(stats2, 128);

        edge_msg_kernel<<<EDGE_B, 256, 0, stream>>>(x, eidx, eattr, wh, bfv, bsv, agg, l);

        bn_stats_kernel<<<STAT_B, 256, 0, stream>>>(agg, stats1);
        apply_cbn_res_kernel<<<EB, 256, 0, stream>>>(agg, x, stats1,
                                                     g_cbn + l * HID, b_cbn + l * HID);
        bn_stats_kernel<<<STAT_B, 256, 0, stream>>>(agg, stats2);
        apply_obn_sp_kernel<<<EB, 256, 0, stream>>>(agg, x, stats2,
                                                    g_obn + l * HID, b_obn + l * HID);
    }

    zerof_kernel<<<((N_GRAPHS * HID + N_GRAPHS) + 255) / 256, 256, 0, stream>>>(pooled, (long)N_GRAPHS * HID + N_GRAPHS);
    pool_kernel<<<EB, 256, 0, stream>>>(x, batch_ids, pooled, cnts);
    mlp_kernel<<<(N_GRAPHS + 255) / 256, 256, 0, stream>>>(pooled, cnts, W1, b1, W2, b2, out);
}
